// DendriteAttention_63067299774999
// MI455X (gfx1250) — matrix operand order repaired and checked by reading, not compiled, not run on hardware
//
#include <hip/hip_runtime.h>

// ---------------------------------------------------------------------------
// Types
// ---------------------------------------------------------------------------
typedef __attribute__((ext_vector_type(16))) __bf16          v16bf;
typedef __attribute__((ext_vector_type(8)))  float           v8f;
typedef __attribute__((ext_vector_type(4)))  float           f32x4;
typedef __attribute__((ext_vector_type(4)))  unsigned short  u16x4;
typedef __attribute__((ext_vector_type(8)))  unsigned short  u16x8;
typedef __attribute__((ext_vector_type(16))) unsigned short  u16x16;

#define BATCH 2
#define SEQ   2048
#define HID   2048
#define NHEAD 16
#define HDIM  128

#define CAT16BF(a, b) __builtin_bit_cast(v16bf, __builtin_shufflevector((a), (b), \
    0, 1, 2, 3, 4, 5, 6, 7, 8, 9, 10, 11, 12, 13, 14, 15))

__device__ __forceinline__ unsigned short f32_to_bf16(float f) {
  unsigned int u = __builtin_bit_cast(unsigned int, f);
  unsigned int r = u + 0x7fffu + ((u >> 16) & 1u);  // round-to-nearest-even
  return (unsigned short)(r >> 16);
}
__device__ __forceinline__ float bf16_to_f32(unsigned short h) {
  unsigned int u = ((unsigned int)h) << 16;
  return __builtin_bit_cast(float, u);
}

// A-fragment (16x32 bf16): lane m=L%16 reads 16 consecutive K at 16*(L/16)
__device__ __forceinline__ v16bf load_afrag(const unsigned short* p) {
  return __builtin_bit_cast(v16bf, *(const u16x16*)p);
}
// B-fragment (32x16 bf16): lane n=L%16 reads 16 consecutive K at 16*(L/16)
__device__ __forceinline__ v16bf load_bfrag(const unsigned short* p) {
  return __builtin_bit_cast(v16bf, *(const u16x16*)p);
}

// ---------------------------------------------------------------------------
// Pass 0: fp32 -> bf16 conversion (bandwidth-bound, vectorized)
// ---------------------------------------------------------------------------
__global__ void __launch_bounds__(256)
cvt_f32_bf16(const float* __restrict__ src, unsigned short* __restrict__ dst, int n4) {
  int i = blockIdx.x * blockDim.x + threadIdx.x;
  if (i >= n4) return;
  f32x4 v = ((const f32x4*)src)[i];
  u16x4 o;
  o.x = f32_to_bf16(v.x);
  o.y = f32_to_bf16(v.y);
  o.z = f32_to_bf16(v.z);
  o.w = f32_to_bf16(v.w);
  ((u16x4*)dst)[i] = o;
}

// ---------------------------------------------------------------------------
// Pass 1: QKV projection GEMM, out = X @ W^T + b  (bf16 in, f32 acc, bf16 out)
// One wave computes a 32(M) x 64(N) tile; K loop unrolled x2 with ping-pong
// register buffers (no copy-rotation, no WMMA source WAR hazards).
// blockIdx.z selects matrix (0=Q, 1=K, 2=V). Q/K written [B,h,S,d];
// V written transposed [B,h,d,S] so the PV matmul loads are contiguous.
// ---------------------------------------------------------------------------
__global__ void __launch_bounds__(256)
qkv_gemm(const unsigned short* __restrict__ Xb,
         const unsigned short* __restrict__ W0,
         const unsigned short* __restrict__ W1,
         const unsigned short* __restrict__ W2,
         const float* __restrict__ b0,
         const float* __restrict__ b1,
         const float* __restrict__ b2,
         unsigned short* __restrict__ Qb,
         unsigned short* __restrict__ Kb,
         unsigned short* __restrict__ Vt) {
  const int mat = blockIdx.z;
  const unsigned short* W = (mat == 0) ? W0 : ((mat == 1) ? W1 : W2);
  const float* bias = (mat == 0) ? b0 : ((mat == 1) ? b1 : b2);

  const int lane = threadIdx.x & 31;
  const int wid  = (blockIdx.x * blockDim.x + threadIdx.x) >> 5;  // 0..4095
  const int mrow = wid >> 5;   // 0..127  (M tiles of 32)
  const int ncol = wid & 31;   // 0..31   (N tiles of 64)
  const int m0 = mrow * 32;
  const int n0 = ncol * 64;
  const int half = lane >> 4;
  const int lm   = lane & 15;

  v8f acc[2][4] = {};  // [M-subtile][N-subtile]

  const unsigned short* arow0 = Xb + (long long)(m0 + lm) * HID + half * 16;
  const unsigned short* arow1 = arow0 + 16LL * HID;
  const unsigned short* brow  = W + (long long)(n0 + lm) * HID + half * 16;

  v16bf aA[2], bA[4], aB[2], bB[4];

  // One pipeline step: consume (a_use,b_use) for K-chunk kk, preload
  // (a_load,b_load) for kk+32 (guarded).
  auto gemm_step = [&](int kk, v16bf (&a_use)[2], v16bf (&b_use)[4],
                       v16bf (&a_load)[2], v16bf (&b_load)[4]) {
    int kn = kk + 32;
    if (kn < HID) {
      a_load[0] = load_afrag(arow0 + kn);
      a_load[1] = load_afrag(arow1 + kn);
#pragma unroll
      for (int nt = 0; nt < 4; ++nt)
        b_load[nt] = load_bfrag(brow + (long long)nt * 16 * HID + kn);
    }
#pragma unroll
    for (int nt = 0; nt < 4; ++nt) {
#pragma unroll
      for (int mi = 0; mi < 2; ++mi) {
        acc[mi][nt] = __builtin_amdgcn_wmma_f32_16x16x32_bf16(
            false, a_use[mi], false, b_use[nt], (short)0, acc[mi][nt], false, false);
      }
    }
  };

  // Prologue: load chunk 0 into buffer A
  aA[0] = load_afrag(arow0);
  aA[1] = load_afrag(arow1);
#pragma unroll
  for (int nt = 0; nt < 4; ++nt) bA[nt] = load_bfrag(brow + (long long)nt * 16 * HID);

#pragma unroll 1
  for (int kk = 0; kk < HID; kk += 64) {   // HID/32 = 64 chunks -> 32 double-steps
    gemm_step(kk,      aA, bA, aB, bB);    // use A, load B
    gemm_step(kk + 32, aB, bB, aA, bA);    // use B, load A
  }

#pragma unroll
  for (int nt = 0; nt < 4; ++nt) {
    int col = n0 + nt * 16 + lm;
    float bs = bias[col];
    int head = col >> 7;
    int dim  = col & (HDIM - 1);
#pragma unroll
    for (int mi = 0; mi < 2; ++mi) {
#pragma unroll
      for (int r = 0; r < 8; ++r) {
        int row = m0 + mi * 16 + r + half * 8;
        int bb = row >> 11;
        int s  = row & (SEQ - 1);
        unsigned short h16 = f32_to_bf16(acc[mi][nt][r] + bs);
        if (mat == 0)
          Qb[((long long)(bb * NHEAD + head) * SEQ + s) * HDIM + dim] = h16;
        else if (mat == 1)
          Kb[((long long)(bb * NHEAD + head) * SEQ + s) * HDIM + dim] = h16;
        else
          Vt[((long long)(bb * NHEAD + head) * HDIM + dim) * SEQ + s] = h16;
      }
    }
  }
}

// ---------------------------------------------------------------------------
// Pass 2: RoPE on Q and K in-place (bf16). Folds softmax 1/sqrt(d) into Q.
// ---------------------------------------------------------------------------
__global__ void __launch_bounds__(256)
rope_kernel(unsigned short* __restrict__ Qb, unsigned short* __restrict__ Kb) {
  long long idx = (long long)blockIdx.x * blockDim.x + threadIdx.x;
  const long long total = (long long)BATCH * NHEAD * SEQ * (HDIM / 2);
  if (idx >= total) return;
  int i = (int)(idx & 63);      // frequency index 0..63
  long long t = idx >> 6;
  int s = (int)(t & (SEQ - 1)); // position
  long long bh = t >> 11;       // 0..31

  // inv_freq = base^(-i/64) = exp2(-i * log2(10000)/64)
  float inv = exp2f(-(float)i * (13.287712379549449f / 64.0f));
  float ang = (float)s * inv;
  float sn = sinf(ang);
  float cs = cosf(ang);

  long long base = (bh * SEQ + s) * HDIM;
  const float qscale = 0.08838834764831845f;  // 1/sqrt(128)

  float q1 = bf16_to_f32(Qb[base + i]);
  float q2 = bf16_to_f32(Qb[base + i + 64]);
  Qb[base + i]      = f32_to_bf16((q1 * cs - q2 * sn) * qscale);
  Qb[base + i + 64] = f32_to_bf16((q2 * cs + q1 * sn) * qscale);

  float k1 = bf16_to_f32(Kb[base + i]);
  float k2 = bf16_to_f32(Kb[base + i + 64]);
  Kb[base + i]      = f32_to_bf16(k1 * cs - k2 * sn);
  Kb[base + i + 64] = f32_to_bf16(k2 * cs + k1 * sn);
}

// ---------------------------------------------------------------------------
// Pass 3: flash attention, software-pipelined with ping-pong K buffers.
// One wave owns a 16-query tile of one (b,h). Per 32-key step:
//   - issue current V-fragment loads + next step's K-fragment loads
//   - score WMMAs (K frags completed last step -> partial waits)
//   - online softmax (VALU overlaps in-flight loads)
//   - P: C-layout -> LDS -> A-layout; PV WMMAs wait only on V loads.
// ---------------------------------------------------------------------------
__global__ void __launch_bounds__(128)
flash_attn(const unsigned short* __restrict__ Qb,
           const unsigned short* __restrict__ Kb,
           const unsigned short* __restrict__ Vt,
           float* __restrict__ out) {
  __shared__ unsigned short ptile[4][16][32];

  const int lane  = threadIdx.x & 31;
  const int wslot = threadIdx.x >> 5;
  const int wid   = blockIdx.x * 4 + wslot;  // 0..4095
  const int qt = wid & 127;                  // query tile in sequence
  const int bh = wid >> 7;                   // 0..31
  const int b = bh >> 4;
  const int h = bh & (NHEAD - 1);
  const int half = lane >> 4;
  const int lm   = lane & 15;

  const unsigned short* Qh = Qb + (long long)bh * SEQ * HDIM;
  const unsigned short* Kh = Kb + (long long)bh * SEQ * HDIM;
  const unsigned short* Vh = Vt + (long long)bh * HDIM * SEQ;

  const int q0 = qt * 16;

  // Q fragments for the 4 K-chunks of d=128
  v16bf qf[4];
  const unsigned short* qrow = Qh + (long long)(q0 + lm) * HDIM + half * 16;
#pragma unroll
  for (int c = 0; c < 4; ++c) qf[c] = load_afrag(qrow + c * 32);

  v8f ctx[8] = {};
  float mrun[8], lrun[8];
#pragma unroll
  for (int r = 0; r < 8; ++r) { mrun[r] = -1e30f; lrun[r] = 0.0f; }

  v16bf kfA[2][4], kfB[2][4];

  // One pipeline step over keys [k0, k0+32): consume kf_use, preload kf_load
  // for [k0+32, k0+64) (guarded).
  auto attn_step = [&](int k0, v16bf (&kf_use)[2][4], v16bf (&kf_load)[2][4]) {
    // ---- issue current V loads (consumed after softmax)
    v16bf vf[8];
#pragma unroll
    for (int nt = 0; nt < 8; ++nt)
      vf[nt] = load_bfrag(Vh + (long long)(nt * 16 + lm) * SEQ + k0 + half * 16);

    // ---- issue next step's K loads into the other buffer
    int kn = k0 + 32;
    if (kn < SEQ) {
#pragma unroll
      for (int t = 0; t < 2; ++t) {
        const unsigned short* krow =
            Kh + (long long)(kn + t * 16 + lm) * HDIM + half * 16;
#pragma unroll
        for (int c = 0; c < 4; ++c) kf_load[t][c] = load_bfrag(krow + c * 32);
      }
    }
    if (k0 + 64 < SEQ) {  // keep L2 warm two steps ahead
      __builtin_prefetch(Kh + (long long)(k0 + 64 + lm) * HDIM, 0, 0);
      __builtin_prefetch(Vh + (long long)lm * SEQ + k0 + 64, 0, 0);
    }

    // ---- scores: two 16x16 tiles over 32 keys (Q pre-scaled by 1/sqrt d)
    v8f sc[2] = {};
#pragma unroll
    for (int t = 0; t < 2; ++t) {
#pragma unroll
      for (int c = 0; c < 4; ++c) {
        sc[t] = __builtin_amdgcn_wmma_f32_16x16x32_bf16(
            false, qf[c], false, kf_use[t][c], (short)0, sc[t], false, false);
      }
    }

    // ---- online softmax (rows r+8*half; reductions within 16-lane groups)
    float p0v[8], p1v[8], sclv[8];
#pragma unroll
    for (int r = 0; r < 8; ++r) {
      float mx = fmaxf(sc[0][r], sc[1][r]);
      mx = fmaxf(mx, __shfl_xor(mx, 8, 32));
      mx = fmaxf(mx, __shfl_xor(mx, 4, 32));
      mx = fmaxf(mx, __shfl_xor(mx, 2, 32));
      mx = fmaxf(mx, __shfl_xor(mx, 1, 32));
      float mnew = fmaxf(mrun[r], mx);
      float scale = __expf(mrun[r] - mnew);
      float p0 = __expf(sc[0][r] - mnew);
      float p1 = __expf(sc[1][r] - mnew);
      float sum = p0 + p1;
      sum += __shfl_xor(sum, 8, 32);
      sum += __shfl_xor(sum, 4, 32);
      sum += __shfl_xor(sum, 2, 32);
      sum += __shfl_xor(sum, 1, 32);
      lrun[r] = lrun[r] * scale + sum;
      mrun[r] = mnew;
      sclv[r] = scale;
      p0v[r] = p0;
      p1v[r] = p1;
    }
#pragma unroll
    for (int nt = 0; nt < 8; ++nt)
#pragma unroll
      for (int r = 0; r < 8; ++r) ctx[nt][r] *= sclv[r];

    // ---- P: C-layout -> LDS (row-major 16x32 bf16) -> A-layout fragment
#pragma unroll
    for (int r = 0; r < 8; ++r) {
      ptile[wslot][r + half * 8][lm]      = f32_to_bf16(p0v[r]);
      ptile[wslot][r + half * 8][16 + lm] = f32_to_bf16(p1v[r]);
    }
    u16x8 pa = *(const u16x8*)&ptile[wslot][lm][half * 16];
    u16x8 pb = *(const u16x8*)&ptile[wslot][lm][half * 16 + 8];
    v16bf pf = CAT16BF(pa, pb);

    // ---- ctx += P @ V  (Vt rows are contiguous over keys)
#pragma unroll
    for (int nt = 0; nt < 8; ++nt) {
      ctx[nt] = __builtin_amdgcn_wmma_f32_16x16x32_bf16(
          false, pf, false, vf[nt], (short)0, ctx[nt], false, false);
    }
  };

  // Prologue: K fragments for keys [0,32) into buffer A
#pragma unroll
  for (int t = 0; t < 2; ++t) {
    const unsigned short* krow = Kh + (long long)(t * 16 + lm) * HDIM + half * 16;
#pragma unroll
    for (int c = 0; c < 4; ++c) kfA[t][c] = load_bfrag(krow + c * 32);
  }

#pragma unroll 1
  for (int k0 = 0; k0 < SEQ; k0 += 64) {  // SEQ/32 = 64 steps -> 32 double-steps
    attn_step(k0,      kfA, kfB);
    attn_step(k0 + 32, kfB, kfA);
  }

  // ---- normalize and write fp32 output [B,S,H]
#pragma unroll
  for (int nt = 0; nt < 8; ++nt) {
#pragma unroll
    for (int r = 0; r < 8; ++r) {
      int row = q0 + r + half * 8;
      float v = ctx[nt][r] / lrun[r];
      out[((long long)(b * SEQ + row)) * HID + h * HDIM + nt * 16 + lm] = v;
    }
  }
}

// ---------------------------------------------------------------------------
// Launch
// ---------------------------------------------------------------------------
extern "C" void kernel_launch(void* const* d_in, const int* in_sizes, int n_in,
                              void* d_out, int out_size, void* d_ws, size_t ws_size,
                              hipStream_t stream) {
  const float* hs = (const float*)d_in[0];
  const float* Wq = (const float*)d_in[1];
  const float* bq = (const float*)d_in[2];
  const float* Wk = (const float*)d_in[3];
  const float* bk = (const float*)d_in[4];
  const float* Wv = (const float*)d_in[5];
  const float* bv = (const float*)d_in[6];
  float* out = (float*)d_out;

  // Workspace layout (bf16 = ushort), ~92 MB total
  unsigned short* Xb  = (unsigned short*)d_ws;            // 4096*2048
  unsigned short* Wqb = Xb  + 4096LL * 2048;              // 2048*2048
  unsigned short* Wkb = Wqb + 2048LL * 2048;
  unsigned short* Wvb = Wkb + 2048LL * 2048;
  unsigned short* Qb  = Wvb + 2048LL * 2048;              // [B,h,S,d]
  unsigned short* Kb  = Qb  + (long long)BATCH * NHEAD * SEQ * HDIM;
  unsigned short* Vt  = Kb  + (long long)BATCH * NHEAD * SEQ * HDIM; // [B,h,d,S]

  // Pass 0: convert activations + weights to bf16
  {
    int n4 = 4096 * 2048 / 4;
    cvt_f32_bf16<<<(n4 + 255) / 256, 256, 0, stream>>>(hs, Xb, n4);
    n4 = 2048 * 2048 / 4;
    cvt_f32_bf16<<<(n4 + 255) / 256, 256, 0, stream>>>(Wq, Wqb, n4);
    cvt_f32_bf16<<<(n4 + 255) / 256, 256, 0, stream>>>(Wk, Wkb, n4);
    cvt_f32_bf16<<<(n4 + 255) / 256, 256, 0, stream>>>(Wv, Wvb, n4);
  }

  // Pass 1: QKV projections (z = matrix select); 4096 waves/matrix
  qkv_gemm<<<dim3(512, 1, 3), 256, 0, stream>>>(Xb, Wqb, Wkb, Wvb,
                                                bq, bk, bv, Qb, Kb, Vt);

  // Pass 2: RoPE (+ fold 1/sqrt(d) into Q)
  {
    long long n = (long long)BATCH * NHEAD * SEQ * (HDIM / 2);
    rope_kernel<<<(int)((n + 255) / 256), 256, 0, stream>>>(Qb, Kb);
  }

  // Pass 3: flash attention -> fp32 output
  flash_attn<<<1024, 128, 0, stream>>>(Qb, Kb, Vt, out);
}
